// NormLinearAttention_35433480192023
// MI455X (gfx1250) — compile-verified, matrix operand order checked by reading, not run on hardware
//
#include <hip/hip_runtime.h>
#include <hip/hip_bf16.h>

// NormLinearAttention for MI455X (gfx1250, wave32, WMMA bf16 path).
// Pipeline: [qkv_proj (WMMA, fragment-ordered LDS, K-step 64)] ->
//           [chunked causal linear attention (WMMA + TDM tensor_load_to_lds)]
//           -> [layernorm] -> [out_proj (WMMA)]

typedef __bf16 bf16_t;
typedef __attribute__((ext_vector_type(16))) __bf16 v16bf;
typedef __attribute__((ext_vector_type(8)))  float  v8f;

constexpr int N_B     = 2;
constexpr int L_SEQ   = 2048;
constexpr int D_MODEL = 1024;
constexpr int H_HEADS = 16;
constexpr int HD      = 64;
#define LN_EPS 1e-5f

// ---------------------------------------------------------------------------
// TDM availability / arity detection (5-arg ROCm 7.2 vs 6-arg therock headers)
// ---------------------------------------------------------------------------
#if defined(__has_builtin)
#  if __has_builtin(__builtin_amdgcn_tensor_load_to_lds)
#    define USE_TDM 1
#  endif
#endif
#ifndef USE_TDM
#  define USE_TDM 0
#endif

#if USE_TDM
typedef __attribute__((ext_vector_type(4))) unsigned u32x4;
typedef __attribute__((ext_vector_type(4))) int      i32x4;
typedef __attribute__((ext_vector_type(8))) int      i32x8;
#  if __has_include(<hip/amd_detail/amd_gfx1250_TDM.h>)
#    pragma message("CDNA5 probe: TDM ENABLED, 6-arg builtin (therock headers)")
#    define TDM_LOAD(g0, g1, g2, g3, cpol)                                    \
       __builtin_amdgcn_tensor_load_to_lds((g0), (g1), (g2), (g3),            \
                                           (i32x8){0,0,0,0,0,0,0,0}, (cpol))
#  else
#    pragma message("CDNA5 probe: TDM ENABLED, 5-arg builtin (ROCm 7.2)")
#    define TDM_LOAD(g0, g1, g2, g3, cpol)                                    \
       __builtin_amdgcn_tensor_load_to_lds((g0), (g1), (g2), (g3), (cpol))
#  endif
#else
#  pragma message("CDNA5 probe: TDM DISABLED - builtin not available")
#endif

// ---------------------------------------------------------------------------
// WMMA fragment helpers (ISA 7.12.2 layouts, wave32)
// ---------------------------------------------------------------------------

// Row-major source: element (m,k) at base[m*ld + k] (A, or B stored N x K).
__device__ __forceinline__ v16bf frag_row(const bf16_t* base, int ld) {
  const int lane = threadIdx.x & 31;
  const int m    = lane & 15;
  const int kh   = (lane >> 4) << 3;   // 0 or 8
  const bf16_t* p = base + m * ld;
  v16bf f;
#pragma unroll
  for (int j = 0; j < 8; ++j) {
    const int k = ((j < 4) ? 0 : 16) + kh + ((j & 3) << 1);
    f[2 * j]     = p[k];
    f[2 * j + 1] = p[k + 1];
  }
  return f;
}

// K-major source: element (m,k) at base[k*ld + m] (transposed reads).
__device__ __forceinline__ v16bf frag_col(const bf16_t* base, int ld) {
  const int lane = threadIdx.x & 31;
  const int m    = lane & 15;
  const int kh   = (lane >> 4) << 3;
  v16bf f;
#pragma unroll
  for (int j = 0; j < 8; ++j) {
    const int k = ((j < 4) ? 0 : 16) + kh + ((j & 3) << 1);
    f[2 * j]     = base[k * ld + m];
    f[2 * j + 1] = base[(k + 1) * ld + m];
  }
  return f;
}

// Fragment-ordered LDS: each lane's 16 bf16 contiguous -> ds_load_b128 x2.
__device__ __forceinline__ v16bf frag_ld(const bf16_t* strip) {
  return *(const v16bf*)(strip + (threadIdx.x & 31) * 16);
}

// Inverse A-frag layout for f32 staging over a 64-wide K tile (two 32-K halves
// at halfStride apart): float4 at (row, c4) -> 4 consecutive bf16.
__device__ __forceinline__ void stage_f32_frag(bf16_t* tile, int halfStride,
                                               int row, int c4, float4 v4) {
  const int kh2   = c4 >> 5;           // which 32-K half
  const int c     = c4 & 31;
  const int strip = row >> 4;
  const int m     = row & 15;
  const int lane  = m + ((c & 8) ? 16 : 0);
  const int e0    = ((c & 16) ? 8 : 0) + ((c & 4) ? 4 : 0);
  bf16_t* s = tile + kh2 * halfStride + (strip * 32 + lane) * 16 + e0;
  s[0] = (bf16_t)v4.x; s[1] = (bf16_t)v4.y;
  s[2] = (bf16_t)v4.z; s[3] = (bf16_t)v4.w;
}

// Same for 8 consecutive bf16 (k = g..g+7): one 16B LDS store.
__device__ __forceinline__ void stage_bf16_frag(bf16_t* tile, int halfStride,
                                                int row, int g, uint4 raw) {
  const int kh2   = g >> 5;
  const int gg    = g & 31;
  const int strip = row >> 4;
  const int m     = row & 15;
  const int lane  = m + ((gg & 8) ? 16 : 0);
  const int e0    = (gg & 16) ? 8 : 0;
  *(uint4*)(tile + kh2 * halfStride + (strip * 32 + lane) * 16 + e0) = raw;
}

__device__ __forceinline__ v8f wmma_bf16(v16bf a, v16bf b, v8f c) {
  return __builtin_amdgcn_wmma_f32_16x16x32_bf16(
      false, a, false, b, (short)0, c, false, false);
}

// ---------------------------------------------------------------------------
// Kernel 1: fused QKV projection  y = act(x @ W^T + b), split to heads (bf16)
//   grid = (D/64, M/128, 3[q,k,v]); block = 256 (8 waves); K-step 64
// ---------------------------------------------------------------------------
constexpr int MT = 128, NT = 64, KT = 64;
constexpr int AH = MT * 32;   // elems per 32-K half of A tile
constexpr int BH = NT * 32;   // elems per 32-K half of B tile

__global__ __launch_bounds__(256)
void qkv_proj_kernel(const float* __restrict__ X,
                     const float* __restrict__ Wq, const float* __restrict__ bq,
                     const float* __restrict__ Wk, const float* __restrict__ bk,
                     const float* __restrict__ Wv, const float* __restrict__ bv,
                     bf16_t* __restrict__ qh, bf16_t* __restrict__ kh,
                     bf16_t* __restrict__ vh) {
  __shared__ __align__(32) bf16_t As[2 * AH];   // 16 KB, fragment-ordered
  __shared__ __align__(32) bf16_t Bs[2 * BH];   //  8 KB, fragment-ordered

  const int z = blockIdx.z;  // 0=q 1=k 2=v
  const float* W    = (z == 0) ? Wq : (z == 1) ? Wk : Wv;
  const float* bias = (z == 0) ? bq : (z == 1) ? bk : bv;
  bf16_t*      dst  = (z == 0) ? qh : (z == 1) ? kh : vh;

  const int m0  = blockIdx.y * MT;
  const int n0  = blockIdx.x * NT;
  const int tid = threadIdx.x;
  const int wid = tid >> 5;

  v8f acc[4] = {};

  for (int k0 = 0; k0 < D_MODEL; k0 += KT) {
    if (k0 + KT < D_MODEL) {  // prefetch next A k-tile (global_prefetch_b8)
      __builtin_prefetch(X + (size_t)(m0 + (tid >> 1)) * D_MODEL + k0 + KT, 0, 3);
    }
#pragma unroll
    for (int t = 0; t < 8; ++t) {            // stage A: 128x64 f32 -> frag bf16
      const int idx = tid + t * 256;         // 0..2047 float4 groups
      const int row = idx >> 4;
      const int c4  = (idx & 15) << 2;
      const float4 v4 = *(const float4*)(X + (size_t)(m0 + row) * D_MODEL + k0 + c4);
      stage_f32_frag(As, AH, row, c4, v4);
    }
#pragma unroll
    for (int t = 0; t < 4; ++t) {            // stage B: W rows are the NxK operand
      const int idx = tid + t * 256;         // 0..1023
      const int row = idx >> 4;
      const int c4  = (idx & 15) << 2;
      const float4 v4 = *(const float4*)(W + (size_t)(n0 + row) * D_MODEL + k0 + c4);
      stage_f32_frag(Bs, BH, row, c4, v4);
    }
    __syncthreads();

#pragma unroll
    for (int kh2 = 0; kh2 < 2; ++kh2) {      // 8 WMMAs per barrier pair
      const v16bf a = frag_ld(As + kh2 * AH + wid * (32 * 16));
#pragma unroll
      for (int cb = 0; cb < 4; ++cb) {
        const v16bf b = frag_ld(Bs + kh2 * BH + cb * (32 * 16));
        acc[cb] = wmma_bf16(a, b, acc[cb]);
      }
    }
    __syncthreads();
  }

  // Epilogue: bias, 1+elu for q/k, scatter to head-major bf16 [n,h,l,hd]
  const int lane  = tid & 31;
  const int colb  = lane & 15;
  const int rbase = (lane >> 4) << 3;
#pragma unroll
  for (int cb = 0; cb < 4; ++cb) {
    const int   cg = n0 + cb * 16 + colb;
    const float bb = bias[cg];
    const int   hh = cg >> 6;
    const int   hd = cg & 63;
#pragma unroll
    for (int r = 0; r < 8; ++r) {
      const int mg = m0 + wid * 16 + rbase + r;
      float v = acc[cb][r] + bb;
      if (z < 2) v = (v > 0.f) ? (v + 1.f) : __expf(v);  // 1+elu
      const int nb = mg >> 11;
      const int ll = mg & (L_SEQ - 1);
      dst[(((size_t)(nb * H_HEADS + hh)) * L_SEQ + ll) * HD + hd] = (bf16_t)v;
    }
  }
}

// ---------------------------------------------------------------------------
// Kernel 2: chunked causal linear attention, one workgroup per (n,h).
//   o_chunk = tril(q k^T) v + q @ S ;  S += k^T v   (S: 64x64 KV state)
//   Chunk tiles arrive via TDM (tensor_load_to_lds) with descriptor padding
//   reproducing the 66-bf16 LDS row stride. grid = (H, N); block = 128.
// ---------------------------------------------------------------------------
constexpr int CH     = 64;
constexpr int LDS_ST = CH + 2;  // 32 dwords data + 1 dword pad per row

#if USE_TDM
__device__ __forceinline__ void tdm_load_chunk(unsigned lds_off,
                                               const bf16_t* gsrc) {
  const unsigned long long ga = (unsigned long long)(uintptr_t)gsrc;
  u32x4 g0;
  g0.x = 1u;                                       // count=1, user descriptor
  g0.y = lds_off;                                  // lds_addr (bytes)
  g0.z = (unsigned)ga;                             // global_addr[31:0]
  g0.w = (unsigned)((ga >> 32) & 0x01ffffffull)    // global_addr[56:32]
         | 0x80000000u;                            // type=2 ("image")
  i32x8 g1;
  g1[0] = (1 << 16)                                // data_size = 2 bytes
        | (1 << 20)                                // pad_enable
        | (4 << 22);                               // pad_interval = 32 dwords
                                                   // pad_amount=0 -> +1 dword
  g1[1] = (CH << 16);                              // tensor_dim0 = 64
  g1[2] = (CH << 16);                              // tensor_dim1 = 64
  g1[3] = (CH << 16);                              // tile_dim0   = 64
  g1[4] = CH;                                      // tile_dim1   = 64
  g1[5] = CH;                                      // tensor_dim0_stride = 64
  g1[6] = 0;
  g1[7] = 0;
  const i32x4 gz = {0, 0, 0, 0};
  TDM_LOAD(g0, g1, gz, gz, 0);
}
#endif

__global__ __launch_bounds__(128)
void linattn_kernel(const bf16_t* __restrict__ qh, const bf16_t* __restrict__ kh,
                    const bf16_t* __restrict__ vh, float* __restrict__ o_att) {
  __shared__ __align__(16) bf16_t qs[CH * LDS_ST], ks[CH * LDS_ST], vs[CH * LDS_ST];
  __shared__ __align__(16) bf16_t wbf[CH * LDS_ST];  // masked intra scores
  __shared__ __align__(16) bf16_t sbf[CH * LDS_ST];  // bf16 mirror of state S

  const int h = blockIdx.x, n = blockIdx.y;
  const int tid = threadIdx.x, wid = tid >> 5, lane = tid & 31;
  const size_t head_base = ((size_t)(n * H_HEADS + h)) * L_SEQ * HD;

  v8f s_acc[4] = {};
  for (int i = tid; i < CH * LDS_ST; i += 128) sbf[i] = (bf16_t)0.f;
  __syncthreads();

  const int colb  = lane & 15;
  const int rbase = (lane >> 4) << 3;

  for (int c = 0; c < L_SEQ / CH; ++c) {
    const size_t cb0 = head_base + (size_t)c * CH * HD;

#if USE_TDM
    // DMA the three 64x64 bf16 tiles into LDS; waves 0..2 issue one each.
    if (wid == 0)      tdm_load_chunk((unsigned)(uintptr_t)qs, qh + cb0);
    else if (wid == 1) tdm_load_chunk((unsigned)(uintptr_t)ks, kh + cb0);
    else if (wid == 2) tdm_load_chunk((unsigned)(uintptr_t)vs, vh + cb0);
    __builtin_amdgcn_s_wait_tensorcnt(0);
#else
#pragma unroll
    for (int t = 0; t < 4; ++t) {
      const int idx = tid + t * 128;
      const int row = idx >> 3;
      const int g   = (idx & 7) << 3;
      union U { uint4 u; bf16_t e[8]; } a, b, v;
      a.u = *(const uint4*)(qh + cb0 + row * HD + g);
      b.u = *(const uint4*)(kh + cb0 + row * HD + g);
      v.u = *(const uint4*)(vh + cb0 + row * HD + g);
#pragma unroll
      for (int e = 0; e < 8; ++e) {
        qs[row * LDS_ST + g + e] = a.e[e];
        ks[row * LDS_ST + g + e] = b.e[e];
        vs[row * LDS_ST + g + e] = v.e[e];
      }
    }
#endif
    __syncthreads();

    // q fragments are the A operand of BOTH the score and q@S matmuls: load once.
    const v16bf aq0 = frag_row(qs + (wid * 16) * LDS_ST + 0,  LDS_ST);
    const v16bf aq1 = frag_row(qs + (wid * 16) * LDS_ST + 32, LDS_ST);

    // (a) intra-chunk scores sc[i][j] = sum_d q[i,d] k[j,d]
    v8f sc[4] = {};
#pragma unroll
    for (int cbk = 0; cbk < 4; ++cbk)
      sc[cbk] = wmma_bf16(aq0, frag_row(ks + (cbk * 16) * LDS_ST + 0, LDS_ST), sc[cbk]);
#pragma unroll
    for (int cbk = 0; cbk < 4; ++cbk)
      sc[cbk] = wmma_bf16(aq1, frag_row(ks + (cbk * 16) * LDS_ST + 32, LDS_ST), sc[cbk]);

    // (b) inter-chunk first: o = q @ S_prev (reads old sbf)
    v8f oc[4] = {};
#pragma unroll
    for (int db = 0; db < 4; ++db)
      oc[db] = wmma_bf16(aq0, frag_col(sbf + 0 * LDS_ST + db * 16, LDS_ST), oc[db]);
#pragma unroll
    for (int db = 0; db < 4; ++db)
      oc[db] = wmma_bf16(aq1, frag_col(sbf + 32 * LDS_ST + db * 16, LDS_ST), oc[db]);

    // (c) causal mask (j <= i), park scores as bf16 A-operand
#pragma unroll
    for (int cbk = 0; cbk < 4; ++cbk) {
#pragma unroll
      for (int r = 0; r < 8; ++r) {
        const int i = wid * 16 + rbase + r;
        const int j = cbk * 16 + colb;
        wbf[i * LDS_ST + j] = (bf16_t)((j <= i) ? sc[cbk][r] : 0.f);
      }
    }
    __syncthreads();

    // (d) intra: o += tril(scores) @ v  (v read transposed as Nd x Kj)
#pragma unroll
    for (int kk = 0; kk < CH; kk += 32) {
      const v16bf aw = frag_row(wbf + (wid * 16) * LDS_ST + kk, LDS_ST);
#pragma unroll
      for (int db = 0; db < 4; ++db) {
        const v16bf bv = frag_col(vs + kk * LDS_ST + db * 16, LDS_ST);
        oc[db] = wmma_bf16(aw, bv, oc[db]);
      }
    }

    // (e) state update: S += k^T v
#pragma unroll
    for (int kk = 0; kk < CH; kk += 32) {
      const v16bf akT = frag_col(ks + kk * LDS_ST + wid * 16, LDS_ST);
#pragma unroll
      for (int db = 0; db < 4; ++db) {
        const v16bf bv = frag_col(vs + kk * LDS_ST + db * 16, LDS_ST);
        s_acc[db] = wmma_bf16(akT, bv, s_acc[db]);
      }
    }

    // (f) refresh bf16 mirror of S for next chunk
#pragma unroll
    for (int db = 0; db < 4; ++db) {
#pragma unroll
      for (int r = 0; r < 8; ++r) {
        const int d1 = wid * 16 + rbase + r;
        const int d2 = db * 16 + colb;
        sbf[d1 * LDS_ST + d2] = (bf16_t)s_acc[db][r];
      }
    }

    // (g) write o chunk (fp32, layout [n,l,h,hd]: D contiguous per row)
#pragma unroll
    for (int db = 0; db < 4; ++db) {
      const int d = db * 16 + colb;
#pragma unroll
      for (int r = 0; r < 8; ++r) {
        const int i  = wid * 16 + rbase + r;
        const int lg = c * CH + i;
        o_att[(((size_t)n * L_SEQ + lg) * H_HEADS + h) * HD + d] = oc[db][r];
      }
    }
    __syncthreads();
  }
}

// ---------------------------------------------------------------------------
// Kernel 3: LayerNorm over D, emit bf16. grid = N*L; block = 256
// ---------------------------------------------------------------------------
__global__ __launch_bounds__(256)
void ln_kernel(const float* __restrict__ o_att, const float* __restrict__ g,
               const float* __restrict__ b, bf16_t* __restrict__ o_n) {
  __shared__ float red[256];
  const int tid = threadIdx.x;
  const size_t row = blockIdx.x;
  const float* x = o_att + row * D_MODEL;

  const float4 xv = *(const float4*)(x + tid * 4);
  red[tid] = xv.x + xv.y + xv.z + xv.w;
  __syncthreads();
  for (int off = 128; off > 0; off >>= 1) {
    if (tid < off) red[tid] += red[tid + off];
    __syncthreads();
  }
  const float mu = red[0] * (1.f / D_MODEL);
  __syncthreads();

  const float d0 = xv.x - mu, d1 = xv.y - mu, d2 = xv.z - mu, d3 = xv.w - mu;
  red[tid] = d0 * d0 + d1 * d1 + d2 * d2 + d3 * d3;
  __syncthreads();
  for (int off = 128; off > 0; off >>= 1) {
    if (tid < off) red[tid] += red[tid + off];
    __syncthreads();
  }
  const float rs = rsqrtf(red[0] * (1.f / D_MODEL) + LN_EPS);

  const float4 gv = *(const float4*)(g + tid * 4);
  const float4 bv = *(const float4*)(b + tid * 4);
  bf16_t* o = o_n + row * D_MODEL + tid * 4;
  o[0] = (bf16_t)(d0 * rs * gv.x + bv.x);
  o[1] = (bf16_t)(d1 * rs * gv.y + bv.y);
  o[2] = (bf16_t)(d2 * rs * gv.z + bv.z);
  o[3] = (bf16_t)(d3 * rs * gv.w + bv.w);
}

// ---------------------------------------------------------------------------
// Kernel 4: output projection  out = o_n @ Wo^T + bo  (fp32 out)
//   grid = (D/64, M/128); block = 256; K-step 64
// ---------------------------------------------------------------------------
__global__ __launch_bounds__(256)
void out_proj_kernel(const bf16_t* __restrict__ Xn, const float* __restrict__ Wo,
                     const float* __restrict__ bo, float* __restrict__ out) {
  __shared__ __align__(32) bf16_t As[2 * AH];
  __shared__ __align__(32) bf16_t Bs[2 * BH];

  const int m0  = blockIdx.y * MT;
  const int n0  = blockIdx.x * NT;
  const int tid = threadIdx.x;
  const int wid = tid >> 5;

  v8f acc[4] = {};

  for (int k0 = 0; k0 < D_MODEL; k0 += KT) {
    if (k0 + KT < D_MODEL) {
      __builtin_prefetch(Xn + (size_t)(m0 + (tid >> 1)) * D_MODEL + k0 + KT, 0, 3);
    }
#pragma unroll
    for (int t = 0; t < 4; ++t) {            // stage A: 128x64 bf16, 16B loads
      const int idx = tid + t * 256;         // 0..1023 groups of 8
      const int row = idx >> 3;
      const int g   = (idx & 7) << 3;
      const uint4 raw = *(const uint4*)(Xn + (size_t)(m0 + row) * D_MODEL + k0 + g);
      stage_bf16_frag(As, AH, row, g, raw);
    }
#pragma unroll
    for (int t = 0; t < 4; ++t) {            // stage B: Wo rows f32 -> bf16
      const int idx = tid + t * 256;
      const int row = idx >> 4;
      const int c4  = (idx & 15) << 2;
      const float4 v4 = *(const float4*)(Wo + (size_t)(n0 + row) * D_MODEL + k0 + c4);
      stage_f32_frag(Bs, BH, row, c4, v4);
    }
    __syncthreads();

#pragma unroll
    for (int kh2 = 0; kh2 < 2; ++kh2) {
      const v16bf a = frag_ld(As + kh2 * AH + wid * (32 * 16));
#pragma unroll
      for (int cb = 0; cb < 4; ++cb) {
        const v16bf b = frag_ld(Bs + kh2 * BH + cb * (32 * 16));
        acc[cb] = wmma_bf16(a, b, acc[cb]);
      }
    }
    __syncthreads();
  }

  const int lane  = tid & 31;
  const int colb  = lane & 15;
  const int rbase = (lane >> 4) << 3;
#pragma unroll
  for (int cb = 0; cb < 4; ++cb) {
    const int   cg = n0 + cb * 16 + colb;
    const float bb = bo[cg];
#pragma unroll
    for (int r = 0; r < 8; ++r) {
      const int mg = m0 + wid * 16 + rbase + r;
      out[(size_t)mg * D_MODEL + cg] = acc[cb][r] + bb;
    }
  }
}

// ---------------------------------------------------------------------------
extern "C" void kernel_launch(void* const* d_in, const int* in_sizes, int n_in,
                              void* d_out, int out_size, void* d_ws, size_t ws_size,
                              hipStream_t stream) {
  (void)in_sizes; (void)n_in; (void)out_size; (void)ws_size;
  const float* query = (const float*)d_in[0];
  const float* Wq = (const float*)d_in[1];
  const float* bq = (const float*)d_in[2];
  const float* Wk = (const float*)d_in[3];
  const float* bk = (const float*)d_in[4];
  const float* Wv = (const float*)d_in[5];
  const float* bv = (const float*)d_in[6];
  const float* Wo = (const float*)d_in[7];
  const float* bo = (const float*)d_in[8];
  const float* ln_g = (const float*)d_in[9];
  const float* ln_b = (const float*)d_in[10];
  float* out = (float*)d_out;

  // Workspace carve-up (~48 MB): qh/kh/vh bf16 head-major, o_att f32, o_n bf16
  const size_t elems = (size_t)N_B * H_HEADS * L_SEQ * HD;  // 4M
  bf16_t* qh    = (bf16_t*)d_ws;
  bf16_t* kh    = qh + elems;
  bf16_t* vh    = kh + elems;
  float*  o_att = (float*)(vh + elems);
  bf16_t* o_n   = (bf16_t*)(o_att + elems);

  dim3 g1(D_MODEL / NT, (N_B * L_SEQ) / MT, 3);
  qkv_proj_kernel<<<g1, 256, 0, stream>>>(query, Wq, bq, Wk, bk, Wv, bv, qh, kh, vh);

  dim3 g2(H_HEADS, N_B);
  linattn_kernel<<<g2, 128, 0, stream>>>(qh, kh, vh, o_att);

  ln_kernel<<<N_B * L_SEQ, 256, 0, stream>>>(o_att, ln_g, ln_b, o_n);

  dim3 g4(D_MODEL / NT, (N_B * L_SEQ) / MT);
  out_proj_kernel<<<g4, 256, 0, stream>>>(o_n, Wo, bo, out);
}
